// MegNetLayer_55851754717346
// MI455X (gfx1250) — matrix operand order repaired and checked by reading, not compiled, not run on hardware
//
#include <hip/hip_runtime.h>

typedef __bf16 bf16_t;
typedef __attribute__((ext_vector_type(16))) __bf16 v16bf;
typedef __attribute__((ext_vector_type(8)))  __bf16 v8bf;
typedef __attribute__((ext_vector_type(8)))  float  v8f;

#define N_NODES 50000
#define N_EDGES 1600000
#define N_BATCH 256

// ---------------- WMMA fragment loaders (wave32, 16x16x32 bf16) ----------------
// A (16x32): lane L holds row m=L%16; elements i: k = (L/16)*16 + i
__device__ __forceinline__ v16bf load_a_frag(const bf16_t* row, int h) {
  v8bf lo = *(const v8bf*)(row + h * 16);
  v8bf hi = *(const v8bf*)(row + h * 16 + 8);
  v16bf r;
#pragma unroll
  for (int i = 0; i < 8; ++i) { r[i] = lo[i]; r[i + 8] = hi[i]; }
  return r;
}
// B (32x16): lane L holds col n=L%16; elements i: k = (L/16)*16 + i  (Wt is [n][k])
__device__ __forceinline__ v16bf load_b_frag(const bf16_t* p) {
  v8bf lo = *(const v8bf*)(p);
  v8bf hi = *(const v8bf*)(p + 8);
  v16bf r;
#pragma unroll
  for (int i = 0; i < 8; ++i) { r[i] = lo[i]; r[i + 8] = hi[i]; }
  return r;
}

// ---------------- GEMM layer: [64 x K] (LDS bf16) x [K x NT*16] -> bias+ReLU -> LDS bf16
template <int K, int NT>
__device__ __forceinline__ void gemm_to_lds(const bf16_t* A, int As,
                                            const bf16_t* Wt, const float* bias,
                                            bf16_t* H, int Hs) {
  const int lane = threadIdx.x & 31;
  const int wave = threadIdx.x >> 5;
  const int mslab = wave & 3;
  const int nh = wave >> 2;
  const int h = lane >> 4;
  const int nl = lane & 15;
  constexpr int TPW = NT / 2;
  v8f acc[TPW];
#pragma unroll
  for (int t = 0; t < TPW; ++t)
#pragma unroll
    for (int j = 0; j < 8; ++j) acc[t][j] = 0.0f;
  const bf16_t* arow = A + (mslab * 16 + nl) * As;
#pragma unroll
  for (int ks = 0; ks < K / 32; ++ks) {
    v16bf a = load_a_frag(arow + ks * 32, h);
#pragma unroll
    for (int t = 0; t < TPW; ++t) {
      const int n = (nh * TPW + t) * 16 + nl;
      v16bf b = load_b_frag(Wt + (size_t)n * K + ks * 32 + h * 16);
      acc[t] = __builtin_amdgcn_wmma_f32_16x16x32_bf16(false, a, false, b,
                                                       (short)0, acc[t], false, false);
    }
  }
#pragma unroll
  for (int t = 0; t < TPW; ++t) {
    const int n = (nh * TPW + t) * 16 + nl;
    const float bs = bias[n];
#pragma unroll
    for (int j = 0; j < 8; ++j) {
      const int ml = mslab * 16 + h * 8 + j;
      float v = acc[t][j] + bs;
      v = fmaxf(v, 0.0f);
      H[ml * Hs + n] = (bf16_t)v;
    }
  }
}

// Final layer: [64 x K] x [K x 64] + bias (no relu), result in registers (2 tiles/wave)
template <int K>
__device__ __forceinline__ void gemm_final(const bf16_t* A, int As,
                                           const bf16_t* Wt, const float* bias,
                                           v8f* out) {
  const int lane = threadIdx.x & 31;
  const int wave = threadIdx.x >> 5;
  const int mslab = wave & 3;
  const int nh = wave >> 2;
  const int h = lane >> 4;
  const int nl = lane & 15;
  v8f acc[2];
#pragma unroll
  for (int t = 0; t < 2; ++t)
#pragma unroll
    for (int j = 0; j < 8; ++j) acc[t][j] = 0.0f;
  const bf16_t* arow = A + (mslab * 16 + nl) * As;
#pragma unroll
  for (int ks = 0; ks < K / 32; ++ks) {
    v16bf a = load_a_frag(arow + ks * 32, h);
#pragma unroll
    for (int t = 0; t < 2; ++t) {
      const int n = (nh * 2 + t) * 16 + nl;
      v16bf b = load_b_frag(Wt + (size_t)n * K + ks * 32 + h * 16);
      acc[t] = __builtin_amdgcn_wmma_f32_16x16x32_bf16(false, a, false, b,
                                                       (short)0, acc[t], false, false);
    }
  }
#pragma unroll
  for (int t = 0; t < 2; ++t) {
    const float bs = bias[(nh * 2 + t) * 16 + nl];
#pragma unroll
    for (int j = 0; j < 8; ++j) out[t][j] = acc[t][j] + bs;
  }
}

// ---------------- small helpers ----------------
__device__ __forceinline__ void copy64bf(bf16_t* dst, const bf16_t* src) {
#pragma unroll
  for (int j = 0; j < 8; ++j) *(v8bf*)(dst + j * 8) = *(const v8bf*)(src + j * 8);
}
__device__ __forceinline__ void cvt64(bf16_t* dst, const float* src, float sc) {
#pragma unroll
  for (int j = 0; j < 8; ++j) {
    v8bf o;
#pragma unroll
    for (int q = 0; q < 8; ++q) o[q] = (bf16_t)(src[j * 8 + q] * sc);
    *(v8bf*)(dst + j * 8) = o;
  }
}

// ---------------- Edge MLP: 64 edges/block, in=256 ----------------
__global__ __launch_bounds__(256) void edge_mlp_kernel(
    const bf16_t* __restrict__ xbf, const float* __restrict__ ea,
    const bf16_t* __restrict__ ubf, const int* __restrict__ ei,
    const int* __restrict__ batch,
    const bf16_t* w1t, const float* b1, const bf16_t* w2t, const float* b2,
    const bf16_t* w3t, const float* b3,
    float* __restrict__ e_out, float* __restrict__ nagg,
    float* __restrict__ cnt_row) {
  __shared__ bf16_t sA[64 * 264];  // L1 input (also reused as L2 output)
  __shared__ bf16_t sH[64 * 136];  // L1 output
  __shared__ int sRow[64];
  const int t = threadIdx.x;
  const size_t base = (size_t)blockIdx.x * 64;
  {
    const int i = t >> 2, seg = t & 3;
    const size_t gi = base + i;
    const int row = ei[gi];
    const int col = ei[N_EDGES + gi];
    const int b = batch[row];
    if (seg == 0) {
      sRow[i] = row;
      unsafeAtomicAdd(&cnt_row[row], 1.0f);
    }
    bf16_t* dst = &sA[i * 264 + seg * 64];
    if (seg == 2) {
      const float* s = ea + gi * 64;
      cvt64(dst, s, 1.0f);
    } else {
      const bf16_t* s = (seg == 0)   ? (xbf + (size_t)row * 64)
                        : (seg == 1) ? (xbf + (size_t)col * 64)
                                     : (ubf + (size_t)b * 64);
      copy64bf(dst, s);
    }
  }
  __syncthreads();
  gemm_to_lds<256, 8>(sA, 264, w1t, b1, sH, 136);
  __syncthreads();
  gemm_to_lds<128, 8>(sH, 136, w2t, b2, sA, 136);
  __syncthreads();
  v8f o2[2];
  gemm_final<128>(sA, 136, w3t, b3, o2);
  const int lane = t & 31, wave = t >> 5;
  const int mslab = wave & 3, nh = wave >> 2, h = lane >> 4, nl = lane & 15;
#pragma unroll
  for (int tt = 0; tt < 2; ++tt) {
    const int n = (nh * 2 + tt) * 16 + nl;
#pragma unroll
    for (int j = 0; j < 8; ++j) {
      const int ml = mslab * 16 + h * 8 + j;
      const size_t gi = base + ml;
      const float v = o2[tt][j];
      e_out[gi * 64 + n] = v;
      unsafeAtomicAdd(&nagg[(size_t)sRow[ml] * 64 + n], v);
    }
  }
}

// ---------------- derive per-batch edge sums from per-node sums ----------------
__global__ __launch_bounds__(256) void batch_edge_sum_kernel(
    const float* __restrict__ nagg, const float* __restrict__ cnt_row,
    const int* __restrict__ batch, float* __restrict__ ebsum,
    float* __restrict__ cnt_eb) {
  const int idx = blockIdx.x * 256 + threadIdx.x;
  if (idx >= N_NODES * 64) return;
  const int i = idx >> 6, n = idx & 63;
  const int b = batch[i];
  unsafeAtomicAdd(&ebsum[b * 64 + n], nagg[idx]);
  if (n == 0) unsafeAtomicAdd(&cnt_eb[b], cnt_row[i]);
}

// ---------------- Node MLP: 64 nodes/block, in=192 ----------------
__global__ __launch_bounds__(256) void node_mlp_kernel(
    const float* __restrict__ nagg, const float* __restrict__ cnt_row,
    const bf16_t* __restrict__ xbf, const bf16_t* __restrict__ ubf,
    const int* __restrict__ batch,
    const bf16_t* w1t, const float* b1, const bf16_t* w2t, const float* b2,
    const bf16_t* w3t, const float* b3,
    float* __restrict__ xn_out, float* __restrict__ xbsum,
    float* __restrict__ cnt_nb) {
  __shared__ bf16_t sA[64 * 200];
  __shared__ bf16_t sH[64 * 136];
  __shared__ int sBat[64];
  const int t = threadIdx.x;
  const size_t base = (size_t)blockIdx.x * 64;
  {
    const int i = t >> 2, seg = t & 3;
    const size_t gi = base + i;
    const bool valid = gi < N_NODES;
    const size_t g = valid ? gi : (size_t)(N_NODES - 1);
    const int b = batch[g];
    bf16_t* dst = &sA[i * 200];
    if (seg == 0) {
      sBat[i] = b;
      if (valid) unsafeAtomicAdd(&cnt_nb[b], 1.0f);
      const float sc = 1.0f / fmaxf(cnt_row[g], 1.0f);
      cvt64(dst, nagg + g * 64, sc);
    } else if (seg == 1) {
      copy64bf(dst + 64, xbf + g * 64);
    } else if (seg == 2) {
      copy64bf(dst + 128, ubf + (size_t)b * 64);
    }
  }
  __syncthreads();
  gemm_to_lds<192, 8>(sA, 200, w1t, b1, sH, 136);
  __syncthreads();
  gemm_to_lds<128, 8>(sH, 136, w2t, b2, sA, 136);
  __syncthreads();
  v8f o2[2];
  gemm_final<128>(sA, 136, w3t, b3, o2);
  const int lane = t & 31, wave = t >> 5;
  const int mslab = wave & 3, nh = wave >> 2, h = lane >> 4, nl = lane & 15;
#pragma unroll
  for (int tt = 0; tt < 2; ++tt) {
    const int n = (nh * 2 + tt) * 16 + nl;
#pragma unroll
    for (int j = 0; j < 8; ++j) {
      const int ml = mslab * 16 + h * 8 + j;
      const size_t gi = base + ml;
      if (gi < N_NODES) {
        const float v = o2[tt][j];
        xn_out[gi * 64 + n] = v;
        unsafeAtomicAdd(&xbsum[(size_t)sBat[ml] * 64 + n], v);
      }
    }
  }
}

// ---------------- Global MLP: 64 graphs/block, in=192 ----------------
__global__ __launch_bounds__(256) void global_mlp_kernel(
    const bf16_t* __restrict__ ubf, const float* __restrict__ xbsum,
    const float* __restrict__ cnt_nb, const float* __restrict__ ebsum,
    const float* __restrict__ cnt_eb,
    const bf16_t* w1t, const float* b1, const bf16_t* w2t, const float* b2,
    const bf16_t* w3t, const float* b3, float* __restrict__ un_out) {
  __shared__ bf16_t sA[64 * 200];
  __shared__ bf16_t sH[64 * 136];
  const int t = threadIdx.x;
  const size_t base = (size_t)blockIdx.x * 64;
  {
    const int i = t >> 2, seg = t & 3;
    const size_t g = base + i;
    bf16_t* dst = &sA[i * 200];
    if (seg == 0) {
      copy64bf(dst, ubf + g * 64);
    } else if (seg == 1) {
      const float sc = 1.0f / fmaxf(cnt_nb[g], 1.0f);
      cvt64(dst + 64, xbsum + g * 64, sc);
    } else if (seg == 2) {
      const float sc = 1.0f / fmaxf(cnt_eb[g], 1.0f);
      cvt64(dst + 128, ebsum + g * 64, sc);
    }
  }
  __syncthreads();
  gemm_to_lds<192, 8>(sA, 200, w1t, b1, sH, 136);
  __syncthreads();
  gemm_to_lds<128, 8>(sH, 136, w2t, b2, sA, 136);
  __syncthreads();
  v8f o2[2];
  gemm_final<128>(sA, 136, w3t, b3, o2);
  const int lane = t & 31, wave = t >> 5;
  const int mslab = wave & 3, nh = wave >> 2, h = lane >> 4, nl = lane & 15;
#pragma unroll
  for (int tt = 0; tt < 2; ++tt) {
    const int n = (nh * 2 + tt) * 16 + nl;
#pragma unroll
    for (int j = 0; j < 8; ++j) {
      const int ml = mslab * 16 + h * 8 + j;
      un_out[(base + ml) * 64 + n] = o2[tt][j];
    }
  }
}

// ---------------- prep kernels ----------------
__global__ void zero_kernel(float* __restrict__ p, int n) {
  const int i = blockIdx.x * 256 + threadIdx.x;
  if (i < n) p[i] = 0.0f;
}
__global__ void cvt_bf16_kernel(const float* __restrict__ s, bf16_t* __restrict__ d, int n) {
  const int i = blockIdx.x * 256 + threadIdx.x;
  if (i < n) d[i] = (bf16_t)s[i];
}
// src: [K][Nout] f32 row-major  ->  dst: [Nout][K] bf16 row-major
__global__ void transpose_cvt_kernel(const float* __restrict__ src, bf16_t* __restrict__ dst,
                                     int K, int Nout) {
  const int idx = blockIdx.x * 256 + threadIdx.x;
  if (idx < K * Nout) {
    const int n = idx / K, k = idx % K;
    dst[idx] = (bf16_t)src[(size_t)k * Nout + n];
  }
}

extern "C" void kernel_launch(void* const* d_in, const int* in_sizes, int n_in,
                              void* d_out, int out_size, void* d_ws, size_t ws_size,
                              hipStream_t stream) {
  (void)in_sizes; (void)n_in; (void)out_size; (void)ws_size;
  const float* x  = (const float*)d_in[0];
  const float* ea = (const float*)d_in[1];
  const float* u  = (const float*)d_in[2];
  const int* ei   = (const int*)d_in[3];
  const int* batch = (const int*)d_in[4];
  const float* W[9]  = {(const float*)d_in[5],  (const float*)d_in[7],  (const float*)d_in[9],
                        (const float*)d_in[11], (const float*)d_in[13], (const float*)d_in[15],
                        (const float*)d_in[17], (const float*)d_in[19], (const float*)d_in[21]};
  const float* Bv[9] = {(const float*)d_in[6],  (const float*)d_in[8],  (const float*)d_in[10],
                        (const float*)d_in[12], (const float*)d_in[14], (const float*)d_in[16],
                        (const float*)d_in[18], (const float*)d_in[20], (const float*)d_in[22]};
  const int wK[9] = {256, 128, 128, 192, 128, 128, 192, 128, 128};
  const int wN[9] = {128, 128,  64, 128, 128,  64, 128, 128,  64};
  const int wOff[9] = {0, 32768, 49152, 57344, 81920, 98304, 106496, 131072, 147456};

  char* ws = (char*)d_ws;
  bf16_t* xbf   = (bf16_t*)(ws);                 // N*64 bf16 = 6,400,000 B
  bf16_t* ubf   = (bf16_t*)(ws + 6400000);       // B*64 bf16 = 32,768 B
  bf16_t* wpool = (bf16_t*)(ws + 6432768);       // 155,648 bf16 = 311,296 B
  float* acc    = (float*)(ws + 6744064);
  float* nagg    = acc;             // N*64
  float* cnt_row = acc + 3200000;   // N (padded)
  float* ebsum   = acc + 3250176;   // B*64
  float* cnt_eb  = acc + 3266560;   // B
  float* xbsum   = acc + 3266816;   // B*64
  float* cnt_nb  = acc + 3283200;   // B
  const int accN = 3283456;

  float* xn_out = (float*)d_out;
  float* e_out  = xn_out + (size_t)N_NODES * 64;
  float* un_out = e_out + (size_t)N_EDGES * 64;

  zero_kernel<<<(accN + 255) / 256, 256, 0, stream>>>(acc, accN);
  cvt_bf16_kernel<<<(N_NODES * 64 + 255) / 256, 256, 0, stream>>>(x, xbf, N_NODES * 64);
  cvt_bf16_kernel<<<(N_BATCH * 64 + 255) / 256, 256, 0, stream>>>(u, ubf, N_BATCH * 64);
  for (int i = 0; i < 9; ++i) {
    const int sz = wK[i] * wN[i];
    transpose_cvt_kernel<<<(sz + 255) / 256, 256, 0, stream>>>(W[i], wpool + wOff[i], wK[i], wN[i]);
  }

  edge_mlp_kernel<<<N_EDGES / 64, 256, 0, stream>>>(
      xbf, ea, ubf, ei, batch,
      wpool + wOff[0], Bv[0], wpool + wOff[1], Bv[1], wpool + wOff[2], Bv[2],
      e_out, nagg, cnt_row);

  batch_edge_sum_kernel<<<(N_NODES * 64 + 255) / 256, 256, 0, stream>>>(
      nagg, cnt_row, batch, ebsum, cnt_eb);

  node_mlp_kernel<<<(N_NODES + 63) / 64, 256, 0, stream>>>(
      nagg, cnt_row, xbf, ubf, batch,
      wpool + wOff[3], Bv[3], wpool + wOff[4], Bv[4], wpool + wOff[5], Bv[5],
      xn_out, xbsum, cnt_nb);

  global_mlp_kernel<<<N_BATCH / 64, 256, 0, stream>>>(
      ubf, xbsum, cnt_nb, ebsum, cnt_eb,
      wpool + wOff[6], Bv[6], wpool + wOff[7], Bv[7], wpool + wOff[8], Bv[8],
      un_out);
}
